// GATBlock_27109833573060
// MI455X (gfx1250) — compile-verified, matrix operand order checked by reading, not run on hardware
//
#include <hip/hip_runtime.h>
#include <hip/hip_bf16.h>

#define NN   25000   // nodes
#define EE   400000  // edges (before self loops)
#define DIN  1024
#define HH   4
#define CC   256
#define HC   (HH*CC) // 1024
#define NEG_SLOPE 0.2f

typedef __attribute__((ext_vector_type(16))) __bf16 v16bf;
typedef __attribute__((ext_vector_type(8)))  float  v8f;

union FragAB { v16bf v; uint4 q[2]; };

// fp32 -> bf16 round-to-nearest-even
static __device__ __forceinline__ unsigned short f2bf(float f) {
    unsigned u = __float_as_uint(f);
    u += 0x7FFFu + ((u >> 16) & 1u);
    return (unsigned short)(u >> 16);
}

static __device__ __forceinline__ unsigned pack2bf(float a, float b) {
    return (unsigned)f2bf(a) | ((unsigned)f2bf(b) << 16);
}

static __device__ __forceinline__ void atomicMaxF32(float* addr, float val) {
    if (val >= 0.0f) {
        atomicMax((int*)addr, __float_as_int(val));
    } else {
        atomicMin((unsigned int*)addr, __float_as_uint(val));
    }
}

static __device__ __forceinline__ void atomicAddF32(float* addr, float val) {
    unsafeAtomicAdd(addr, val);   // lowers to global_atomic_add_f32
}

// CDNA5 async copy global -> LDS (ASYNCcnt-tracked, no VGPR round trip)
static __device__ __forceinline__ void async_load_b128(unsigned lds_off,
                                                       const void* gaddr) {
    asm volatile("global_load_async_to_lds_b128 %0, %1, off"
                 :: "v"(lds_off), "v"((unsigned long long)(size_t)gaddr)
                 : "memory");
}
static __device__ __forceinline__ void wait_asynccnt0() {
    asm volatile("s_wait_asynccnt 0x0" ::: "memory");
}

// ---------------------------------------------------------------------------
// Init: zero output accumulator, m = -inf, z = 0
// ---------------------------------------------------------------------------
__global__ void __launch_bounds__(256) gat_init(float* __restrict__ out,
                                                float* __restrict__ m,
                                                float* __restrict__ z) {
    int gid = blockIdx.x * 256 + threadIdx.x;
    if (gid < NN * HC) out[gid] = 0.0f;
    if (gid < NN * HH) { m[gid] = -__builtin_inff(); z[gid] = 0.0f; }
}

// ---------------------------------------------------------------------------
// Convert x (fp32) -> xbf (bf16), 8 elements per thread
// ---------------------------------------------------------------------------
__global__ void __launch_bounds__(256) conv_x_bf16(const float* __restrict__ x,
                                                   unsigned short* __restrict__ xbf) {
    int gid = blockIdx.x * 256 + threadIdx.x;
    if (gid >= NN * DIN / 8) return;
    const float4* p = (const float4*)x + (size_t)gid * 2;
    float4 v0 = p[0], v1 = p[1];
    uint4 o;
    o.x = pack2bf(v0.x, v0.y);
    o.y = pack2bf(v0.z, v0.w);
    o.z = pack2bf(v1.x, v1.y);
    o.w = pack2bf(v1.z, v1.w);
    ((uint4*)xbf)[gid] = o;
}

// ---------------------------------------------------------------------------
// Convert + transpose W[k][c] (fp32) -> Wt[c][k] (bf16), 64x64 LDS tiles
// ---------------------------------------------------------------------------
__global__ void __launch_bounds__(256) conv_w_bf16t(const float* __restrict__ W,
                                                    unsigned short* __restrict__ Wt) {
    __shared__ unsigned short tile[64][65];
    const int c0 = blockIdx.x * 64;   // column tile
    const int k0 = blockIdx.y * 64;   // k tile
    const int tid = threadIdx.x;
#pragma unroll
    for (int i = 0; i < 16; ++i) {
        int lin = tid + i * 256;
        int kk = lin >> 6, cc = lin & 63;
        tile[cc][kk] = f2bf(W[(size_t)(k0 + kk) * HC + c0 + cc]);
    }
    __syncthreads();
#pragma unroll
    for (int i = 0; i < 16; ++i) {
        int lin = tid + i * 256;
        int rr = lin >> 6, kk = lin & 63;
        Wt[(size_t)(c0 + rr) * DIN + k0 + kk] = tile[rr][kk];
    }
}

// ---------------------------------------------------------------------------
// Projection GEMM: xp[N, HC] = x[N, DIN] @ W[DIN, HC], bf16 inputs via
// async global->LDS copies, f32 WMMA accumulate.
// Block tile 128(M) x 128(N), K-step 32, 8 waves, each wave 2x4 WMMA tiles.
// A rows are CLAMPED during staging: out-of-range A rows only influence
// out-of-range C rows, which are never stored.
// ---------------------------------------------------------------------------
#define BM 128
#define BN 128
#define BK 32
#define LDA 40   // 32 + 8 pad shorts -> 80B rows, 16B aligned
#define LDB 40

__global__ void __launch_bounds__(256) gat_gemm(const unsigned short* __restrict__ xbf,
                                                const unsigned short* __restrict__ Wt,
                                                float* __restrict__ xp) {
    __shared__ __align__(16) unsigned short As[BM * LDA];
    __shared__ __align__(16) unsigned short Bs[BN * LDB];

    const int tid  = threadIdx.x;
    const int m0   = blockIdx.x * BM;
    const int n0   = blockIdx.y * BN;
    const int wave = tid >> 5;
    const int lane = tid & 31;
    const int wm   = wave >> 1;        // 0..3  -> 32-row strip
    const int wn   = wave & 1;         // 0..1  -> 64-col strip
    const int lhalf = lane >> 4;       // 0/1
    const int l15   = lane & 15;
    const int kbase = lhalf * 8;       // ISA 16-bit A/B fragment K grouping

    // LDS byte offsets (generic pointer low 32 bits == LDS address)
    const unsigned asBase = (unsigned)(size_t)&As[0];
    const unsigned bsBase = (unsigned)(size_t)&Bs[0];

    v8f acc[2][4] = {};

    for (int k0 = 0; k0 < DIN; k0 += BK) {
        // ---- async stage A (128x32) and B (128 cols x 32 k), 16B chunks ----
#pragma unroll
        for (int i = 0; i < 2; ++i) {
            int q  = tid + i * 256;        // 0..511 chunk id
            int r  = q >> 2;               // row / column
            int c8 = q & 3;                // 8-element k group
            int grow = m0 + r;
            grow = (grow < NN) ? grow : (NN - 1);
            async_load_b128(asBase + 80u * r + 16u * c8,
                            xbf + (size_t)grow * DIN + k0 + c8 * 8);
            async_load_b128(bsBase + 80u * r + 16u * c8,
                            Wt + (size_t)(n0 + r) * DIN + k0 + c8 * 8);
        }
        wait_asynccnt0();
        __syncthreads();

        FragAB a[2], b[4];
#pragma unroll
        for (int mt = 0; mt < 2; ++mt) {
            const unsigned short* p = &As[(wm * 32 + mt * 16 + l15) * LDA + kbase];
            a[mt].q[0] = *(const uint4*)(p);
            a[mt].q[1] = *(const uint4*)(p + 16);
        }
#pragma unroll
        for (int nt = 0; nt < 4; ++nt) {
            const unsigned short* p = &Bs[(wn * 64 + nt * 16 + l15) * LDB + kbase];
            b[nt].q[0] = *(const uint4*)(p);
            b[nt].q[1] = *(const uint4*)(p + 16);
        }
#pragma unroll
        for (int mt = 0; mt < 2; ++mt)
#pragma unroll
            for (int nt = 0; nt < 4; ++nt)
                acc[mt][nt] = __builtin_amdgcn_wmma_f32_16x16x32_bf16(
                    false, a[mt].v, false, b[nt].v,
                    (short)0, acc[mt][nt], false, false);
        __syncthreads();
    }

    // Store D: VGPR j -> row j + 8*lhalf, col = lane&15 (per ISA C/D layout)
#define STORE_C(GUARDED)                                                      \
    _Pragma("unroll")                                                         \
    for (int mt = 0; mt < 2; ++mt) {                                          \
        _Pragma("unroll")                                                     \
        for (int nt = 0; nt < 4; ++nt) {                                      \
            int col = n0 + wn * 64 + nt * 16 + l15;                           \
            _Pragma("unroll")                                                 \
            for (int j = 0; j < 8; ++j) {                                     \
                int row = m0 + wm * 32 + mt * 16 + lhalf * 8 + j;             \
                if (!GUARDED || row < NN)                                     \
                    xp[(size_t)row * HC + col] = acc[mt][nt][j];              \
            }                                                                 \
        }                                                                     \
    }

    if (m0 + BM <= NN) {
        STORE_C(false)
    } else {
        STORE_C(true)
    }
#undef STORE_C
}

// ---------------------------------------------------------------------------
// Per-node attention logits: a_src[n,h] = <xp[n,h,:], att_src[h,:]>, same dst
// ---------------------------------------------------------------------------
__global__ void __launch_bounds__(256) gat_att(const float* __restrict__ xp,
                                               const float* __restrict__ att_src,
                                               const float* __restrict__ att_dst,
                                               float* __restrict__ a_src,
                                               float* __restrict__ a_dst) {
    int gid = blockIdx.x * 256 + threadIdx.x;
    if (gid >= NN * HH) return;
    int n = gid >> 2, h = gid & 3;
    const float* xr = xp + (size_t)n * HC + h * CC;
    const float* as = att_src + h * CC;
    const float* ad = att_dst + h * CC;
    float s = 0.0f, d = 0.0f;
#pragma unroll 4
    for (int c = 0; c < CC; c += 4) {
        float4 xv = *(const float4*)(xr + c);
        float4 av = *(const float4*)(as + c);
        float4 dv = *(const float4*)(ad + c);
        s += xv.x * av.x + xv.y * av.y + xv.z * av.z + xv.w * av.w;
        d += xv.x * dv.x + xv.y * dv.y + xv.z * dv.z + xv.w * dv.w;
    }
    a_src[gid] = s;
    a_dst[gid] = d;
}

// ---------------------------------------------------------------------------
// Edge pass 1: e = leaky_relu(a_src[s]+a_dst[d]); segment max into m[dst]
// Edge idx in [0, E+N); idx >= E are self loops.
// ---------------------------------------------------------------------------
__global__ void __launch_bounds__(256) gat_edge_max(const int* __restrict__ esrc,
                                                    const int* __restrict__ edst,
                                                    const float* __restrict__ a_src,
                                                    const float* __restrict__ a_dst,
                                                    float* __restrict__ m) {
    int idx = blockIdx.x * 256 + threadIdx.x;
    if (idx >= EE + NN) return;
    int s = (idx < EE) ? esrc[idx] : (idx - EE);
    int d = (idx < EE) ? edst[idx] : (idx - EE);
#pragma unroll
    for (int h = 0; h < HH; ++h) {
        float e = a_src[s * HH + h] + a_dst[d * HH + h];
        e = (e > 0.0f) ? e : NEG_SLOPE * e;
        atomicMaxF32(&m[d * HH + h], e);
    }
}

// ---------------------------------------------------------------------------
// Edge pass 2: ee = exp(e - m[dst]); segment sum into z[dst]; stash ee
// ---------------------------------------------------------------------------
__global__ void __launch_bounds__(256) gat_edge_exp(const int* __restrict__ esrc,
                                                    const int* __restrict__ edst,
                                                    const float* __restrict__ a_src,
                                                    const float* __restrict__ a_dst,
                                                    const float* __restrict__ m,
                                                    float* __restrict__ ee,
                                                    float* __restrict__ z) {
    int idx = blockIdx.x * 256 + threadIdx.x;
    if (idx >= EE + NN) return;
    int s = (idx < EE) ? esrc[idx] : (idx - EE);
    int d = (idx < EE) ? edst[idx] : (idx - EE);
#pragma unroll
    for (int h = 0; h < HH; ++h) {
        float e = a_src[s * HH + h] + a_dst[d * HH + h];
        e = (e > 0.0f) ? e : NEG_SLOPE * e;
        float w = __expf(e - m[d * HH + h]);
        ee[idx * HH + h] = w;
        atomicAddF32(&z[d * HH + h], w);
    }
}

// ---------------------------------------------------------------------------
// Edge pass 3: out[dst] += (ee/z[dst]) * xp[src]   (one block per edge)
// thread t covers channels [4t, 4t+4), head h = t/64
// ---------------------------------------------------------------------------
__global__ void __launch_bounds__(256) gat_aggregate(const int* __restrict__ esrc,
                                                     const int* __restrict__ edst,
                                                     const float* __restrict__ ee,
                                                     const float* __restrict__ z,
                                                     const float* __restrict__ xp,
                                                     float* __restrict__ out) {
    int idx = blockIdx.x;
    int s = (idx < EE) ? esrc[idx] : (idx - EE);
    int d = (idx < EE) ? edst[idx] : (idx - EE);
    int t = threadIdx.x;
    int h = t >> 6;
    float alpha = ee[idx * HH + h] / z[d * HH + h];
    float4 xv = *(const float4*)(xp + (size_t)s * HC + t * 4);
    float* op = out + (size_t)d * HC + t * 4;
    atomicAddF32(op + 0, alpha * xv.x);
    atomicAddF32(op + 1, alpha * xv.y);
    atomicAddF32(op + 2, alpha * xv.z);
    atomicAddF32(op + 3, alpha * xv.w);
}

// ---------------------------------------------------------------------------
// Finalize: out = relu(out + bias)
// ---------------------------------------------------------------------------
__global__ void __launch_bounds__(256) gat_final(float* __restrict__ out,
                                                 const float* __restrict__ bias) {
    int gid = blockIdx.x * 256 + threadIdx.x;
    if (gid >= NN * HC) return;
    float v = out[gid] + bias[gid & (HC - 1)];
    out[gid] = (v > 0.0f) ? v : 0.0f;
}

// ---------------------------------------------------------------------------
extern "C" void kernel_launch(void* const* d_in, const int* in_sizes, int n_in,
                              void* d_out, int out_size, void* d_ws, size_t ws_size,
                              hipStream_t stream) {
    const float* x        = (const float*)d_in[0];
    const int*   eidx     = (const int*)  d_in[1];   // [2, E] flat
    const float* W        = (const float*)d_in[2];
    const float* att_src  = (const float*)d_in[3];
    const float* att_dst  = (const float*)d_in[4];
    const float* bias     = (const float*)d_in[5];
    float*       out      = (float*)d_out;

    const int* esrc = eidx;
    const int* edst = eidx + EE;

    // Workspace layout (floats unless noted). The bf16 staging buffers are
    // dead after gat_gemm, so the ee buffer (first used in gat_edge_exp)
    // aliases the same region.
    float* ws    = (float*)d_ws;
    float* xp    = ws;                                 // NN*HC = 25,600,000 f
    float* a_src = xp    + (size_t)NN * HC;            // NN*HH
    float* a_dst = a_src + (size_t)NN * HH;            // NN*HH
    float* m     = a_dst + (size_t)NN * HH;            // NN*HH
    float* z     = m     + (size_t)NN * HH;            // NN*HH
    float* scratch = z + (size_t)NN * HH;              // union region
    unsigned short* xbf = (unsigned short*)scratch;    // NN*DIN  bf16
    unsigned short* Wt  = xbf + (size_t)NN * DIN;      // HC*DIN  bf16 (transposed)
    float* ee    = scratch;                            // (EE+NN)*HH f (aliases xbf)

    const int TOT = EE + NN;

    gat_init<<<(NN * HC + 255) / 256, 256, 0, stream>>>(out, m, z);

    conv_x_bf16<<<(NN * DIN / 8 + 255) / 256, 256, 0, stream>>>(x, xbf);
    dim3 wt(HC / 64, DIN / 64);
    conv_w_bf16t<<<wt, 256, 0, stream>>>(W, Wt);

    dim3 gg((NN + BM - 1) / BM, HC / BN);
    gat_gemm<<<gg, 256, 0, stream>>>(xbf, Wt, xp);

    gat_att<<<(NN * HH + 255) / 256, 256, 0, stream>>>(xp, att_src, att_dst, a_src, a_dst);

    gat_edge_max<<<(TOT + 255) / 256, 256, 0, stream>>>(esrc, edst, a_src, a_dst, m);
    gat_edge_exp<<<(TOT + 255) / 256, 256, 0, stream>>>(esrc, edst, a_src, a_dst, m, ee, z);
    gat_aggregate<<<TOT, 256, 0, stream>>>(esrc, edst, ee, z, xp, out);

    gat_final<<<(NN * HC + 255) / 256, 256, 0, stream>>>(out, bias);
}
